// GIN_41128606826859
// MI455X (gfx1250) — matrix operand order repaired and checked by reading, not compiled, not run on hardware
//
#include <hip/hip_runtime.h>
#include <hip/hip_bf16.h>
#include <stdint.h>

typedef __attribute__((ext_vector_type(16))) _Float16 v16h;
typedef __attribute__((ext_vector_type(8)))  _Float16 v8h;
typedef __attribute__((ext_vector_type(8)))  float    v8f;

#define GIN_N     100000
#define GIN_E     1600000
#define GIN_INC   64
#define GIN_HID   256
#define GIN_OUTC  64
#define GIN_EPS   1e-5f

// ---------------------------------------------------------------------------
// 1) h0 = x  (float4 copy)
// ---------------------------------------------------------------------------
__global__ void gin_copy_f32v4(const float4* __restrict__ x,
                               float4* __restrict__ h0, int n4) {
    int i = blockIdx.x * blockDim.x + threadIdx.x;
    if (i < n4) h0[i] = x[i];
}

// ---------------------------------------------------------------------------
// 2) scatter-add: h0[dst] += x[src] over all edges. 16 threads/edge, 4 ch each.
// ---------------------------------------------------------------------------
__global__ void gin_scatter(const float* __restrict__ x,
                            const long long* __restrict__ ei,
                            float* __restrict__ h0, int E) {
    int t = blockIdx.x * blockDim.x + threadIdx.x;
    int e  = t >> 4;
    int c4 = (t & 15) * 4;
    if (e >= E) return;
    int src = (int)ei[e];
    int dst = (int)ei[E + e];
    float4 v = *(const float4*)(x + (size_t)src * GIN_INC + c4);
    float* d = h0 + (size_t)dst * GIN_INC + c4;
    atomicAdd(d + 0, v.x);
    atomicAdd(d + 1, v.y);
    atomicAdd(d + 2, v.z);
    atomicAdd(d + 3, v.w);
}

// ---------------------------------------------------------------------------
// 3) f32 -> f16 activation convert (vectorized x4)
// ---------------------------------------------------------------------------
__global__ void gin_cvt_f16(const float4* __restrict__ h0,
                            _Float16* __restrict__ h16, int n4) {
    int i = blockIdx.x * blockDim.x + threadIdx.x;
    if (i >= n4) return;
    float4 v = h0[i];
    _Float16* o = h16 + (size_t)i * 4;
    o[0] = (_Float16)v.x; o[1] = (_Float16)v.y;
    o[2] = (_Float16)v.z; o[3] = (_Float16)v.w;
}

// ---------------------------------------------------------------------------
// 4) weight prep: transpose [K,O] f32 -> [O,K] f16 (one-time, tiny)
// ---------------------------------------------------------------------------
__global__ void gin_wprep(const float* __restrict__ w,
                          _Float16* __restrict__ wT, int K, int O) {
    int i = blockIdx.x * blockDim.x + threadIdx.x;
    if (i >= K * O) return;
    int k = i / O, o = i % O;
    wT[(size_t)o * K + k] = (_Float16)w[i];
}

// folded BN: s = g*rsqrt(rv+eps); t = (b - rm)*s + be
__global__ void gin_bnprep(const float* __restrict__ b, const float* __restrict__ g,
                           const float* __restrict__ be, const float* __restrict__ rm,
                           const float* __restrict__ rv,
                           float* __restrict__ s, float* __restrict__ t, int C) {
    int i = blockIdx.x * blockDim.x + threadIdx.x;
    if (i >= C) return;
    float sv = g[i] * rsqrtf(rv[i] + GIN_EPS);
    s[i] = sv;
    t[i] = (b[i] - rm[i]) * sv + be[i];
}

__global__ void gin_b3prep(const float* __restrict__ b3,
                           float* __restrict__ s, float* __restrict__ t, int C) {
    int i = blockIdx.x * blockDim.x + threadIdx.x;
    if (i >= C) return;
    s[i] = 1.0f;
    t[i] = b3[i];
}

// ---------------------------------------------------------------------------
// 5) WMMA GEMM: out[rowTile*16 .. +16, colBase .. +64] =
//      act[N,KTOT](f16) @ wT[COUT,KTOT]^T, then y = acc*s + t (+ReLU),
//      stored as f16 (next layer) or f32 (final output).
//    One wave per 16x64 slab; 4x v8f accumulators; A fragment reused 4x.
// ---------------------------------------------------------------------------
template <int KTOT, int COUT, bool RELU, bool TO_F16>
__global__ void gin_wmma_gemm(const _Float16* __restrict__ act,
                              const _Float16* __restrict__ wT,
                              const float* __restrict__ scale,
                              const float* __restrict__ shift,
                              _Float16* __restrict__ out16,
                              float* __restrict__ out32,
                              int nRowTiles) {
    const int wave = (blockIdx.x * blockDim.x + threadIdx.x) >> 5;
    const int lane = threadIdx.x & 31;
    const int colWaves = COUT / 64;
    const int rowTile  = wave / colWaves;
    const int colBase  = (wave % colWaves) * 64;
    if (rowTile >= nRowTiles) return;   // uniform per-wave: EXEC stays all-ones

    const int half = lane >> 4;   // 0: K-low half, 1: K-high half (per ISA layout)
    const int lr   = lane & 15;
    const int row  = rowTile * 16 + lr;

    const _Float16* aRow = act + (size_t)row * KTOT;
    // warm the weight slab into cache (global_prefetch path)
    __builtin_prefetch(wT + (size_t)(colBase + lr) * KTOT, 0, 1);

    v8f acc[4] = {v8f{}, v8f{}, v8f{}, v8f{}};

#pragma unroll
    for (int k0 = 0; k0 < KTOT; k0 += 32) {
        // A fragment (16x32): lane<16 -> K [k0..k0+7],[k0+16..k0+23]; lane>=16 -> +8
        v8h alo = *(const v8h*)(aRow + k0 + half * 8);
        v8h ahi = *(const v8h*)(aRow + k0 + 16 + half * 8);
        v16h a;
#pragma unroll
        for (int i = 0; i < 8; ++i) { a[i] = alo[i]; a[8 + i] = ahi[i]; }

#pragma unroll
        for (int j = 0; j < 4; ++j) {
            // B fragment (32x16): lane holds column (colBase+j*16+lr),
            // K = k0 + half*8 + [0..7], k0 + 16 + half*8 + [0..7]  (matches A)
            const _Float16* bp =
                wT + (size_t)(colBase + j * 16 + lr) * KTOT + k0 + half * 8;
            v8h blo = *(const v8h*)(bp);
            v8h bhi = *(const v8h*)(bp + 16);
            v16h b;
#pragma unroll
            for (int i = 0; i < 8; ++i) { b[i] = blo[i]; b[8 + i] = bhi[i]; }

            acc[j] = __builtin_amdgcn_wmma_f32_16x16x32_f16(
                false, a, false, b, (short)0, acc[j], false, false);
        }
    }

    // Epilogue: C/D layout -> element (m = r + half*8, n = lr) in acc[j][r]
#pragma unroll
    for (int j = 0; j < 4; ++j) {
        const int ch = colBase + j * 16 + lr;
        const float s = scale[ch];
        const float t = shift[ch];
#pragma unroll
        for (int r = 0; r < 8; ++r) {
            const int m = r + half * 8;
            float v = acc[j][r] * s + t;
            if (RELU) v = fmaxf(v, 0.0f);
            const size_t o = (size_t)(rowTile * 16 + m) * COUT + ch;
            if (TO_F16) out16[o] = (_Float16)v;
            else        out32[o] = v;
        }
    }
}

// ---------------------------------------------------------------------------
// launch
// ---------------------------------------------------------------------------
extern "C" void kernel_launch(void* const* d_in, const int* in_sizes, int n_in,
                              void* d_out, int out_size, void* d_ws, size_t ws_size,
                              hipStream_t stream) {
    const float*     x  = (const float*)d_in[0];
    const long long* ei = (const long long*)d_in[1];
    const float* w1 = (const float*)d_in[2];
    const float* b1 = (const float*)d_in[3];
    const float* g1 = (const float*)d_in[4];
    const float* be1= (const float*)d_in[5];
    const float* rm1= (const float*)d_in[6];
    const float* rv1= (const float*)d_in[7];
    const float* w2 = (const float*)d_in[8];
    const float* b2 = (const float*)d_in[9];
    const float* g2 = (const float*)d_in[10];
    const float* be2= (const float*)d_in[11];
    const float* rm2= (const float*)d_in[12];
    const float* rv2= (const float*)d_in[13];
    const float* w3 = (const float*)d_in[14];
    const float* b3 = (const float*)d_in[15];
    float* out = (float*)d_out;

    // workspace layout (all chunk sizes multiples of 1KB -> aligned)
    char* ws = (char*)d_ws;
    size_t off = 0;
    float*    h0  = (float*)(ws + off);    off += (size_t)GIN_N * GIN_INC * 4;   // 25.6 MB
    _Float16* h16 = (_Float16*)(ws + off); off += (size_t)GIN_N * GIN_INC * 2;   // 12.8 MB
    _Float16* a1  = (_Float16*)(ws + off); off += (size_t)GIN_N * GIN_HID * 2;   // 51.2 MB
    _Float16* a2  = (_Float16*)(ws + off); off += (size_t)GIN_N * GIN_HID * 2;   // 51.2 MB
    _Float16* w1T = (_Float16*)(ws + off); off += (size_t)GIN_INC * GIN_HID * 2;
    _Float16* w2T = (_Float16*)(ws + off); off += (size_t)GIN_HID * GIN_HID * 2;
    _Float16* w3T = (_Float16*)(ws + off); off += (size_t)GIN_HID * GIN_OUTC * 2;
    float* s1 = (float*)(ws + off); off += 1024;
    float* t1 = (float*)(ws + off); off += 1024;
    float* s2 = (float*)(ws + off); off += 1024;
    float* t2 = (float*)(ws + off); off += 1024;
    float* s3 = (float*)(ws + off); off += 1024;
    float* t3 = (float*)(ws + off); off += 1024;

    const int B = 256;

    // 1) h0 = x
    {
        int n4 = GIN_N * GIN_INC / 4;
        gin_copy_f32v4<<<(n4 + B - 1) / B, B, 0, stream>>>((const float4*)x, (float4*)h0, n4);
    }
    // weight/BN prep (independent of scatter; tiny)
    gin_wprep<<<(GIN_INC * GIN_HID + B - 1) / B, B, 0, stream>>>(w1, w1T, GIN_INC, GIN_HID);
    gin_wprep<<<(GIN_HID * GIN_HID + B - 1) / B, B, 0, stream>>>(w2, w2T, GIN_HID, GIN_HID);
    gin_wprep<<<(GIN_HID * GIN_OUTC + B - 1) / B, B, 0, stream>>>(w3, w3T, GIN_HID, GIN_OUTC);
    gin_bnprep<<<1, B, 0, stream>>>(b1, g1, be1, rm1, rv1, s1, t1, GIN_HID);
    gin_bnprep<<<1, B, 0, stream>>>(b2, g2, be2, rm2, rv2, s2, t2, GIN_HID);
    gin_b3prep<<<1, B, 0, stream>>>(b3, s3, t3, GIN_OUTC);

    // 2) scatter-add
    {
        long long tot = (long long)GIN_E * 16;
        gin_scatter<<<(int)((tot + B - 1) / B), B, 0, stream>>>(x, ei, h0, GIN_E);
    }
    // 3) h16 = (f16)h0
    {
        int n4 = GIN_N * GIN_INC / 4;
        gin_cvt_f16<<<(n4 + B - 1) / B, B, 0, stream>>>((const float4*)h0, h16, n4);
    }

    const int rowTiles = GIN_N / 16;  // 6250
    // 4) layer1: [N,64] @ [64,256] -> a1 (f16, BN+ReLU folded)
    {
        int waves = rowTiles * (GIN_HID / 64);
        int blocks = (waves * 32 + B - 1) / B;
        gin_wmma_gemm<GIN_INC, GIN_HID, true, true>
            <<<blocks, B, 0, stream>>>(h16, w1T, s1, t1, a1, nullptr, rowTiles);
    }
    // 5) layer2: [N,256] @ [256,256] -> a2 (f16, BN+ReLU folded)
    {
        int waves = rowTiles * (GIN_HID / 64);
        int blocks = (waves * 32 + B - 1) / B;
        gin_wmma_gemm<GIN_HID, GIN_HID, true, true>
            <<<blocks, B, 0, stream>>>(a1, w2T, s2, t2, a2, nullptr, rowTiles);
    }
    // 6) layer3: [N,256] @ [256,64] + b3 -> out (f32)
    {
        int waves = rowTiles * (GIN_OUTC / 64);
        int blocks = (waves * 32 + B - 1) / B;
        gin_wmma_gemm<GIN_HID, GIN_OUTC, false, false>
            <<<blocks, B, 0, stream>>>(a2, w3T, s3, t3, nullptr, out, rowTiles);
    }
    (void)in_sizes; (void)n_in; (void)out_size; (void)ws_size;
}
